// OCRModel_v20230512_v1_25168508355198
// MI455X (gfx1250) — matrix operand order repaired and checked by reading, not compiled, not run on hardware
//
#include <hip/hip_runtime.h>
#include <hip/hip_bf16.h>

// MI455X / gfx1250 implementation of the FLASH-style (GAU quad+linear attention)
// k-mer model. Heavy matmuls (S=QK^T, O=A*V, lin_kv=K^T*V) run on
// v_wmma_f32_16x16x32_f16 with conflict-free padded-LDS staging and batched
// fragment loads sized to stay within the VGPR budget (no scratch spills).

typedef __attribute__((ext_vector_type(16))) _Float16 v16h;
typedef __attribute__((ext_vector_type(8)))  float    v8f;

#define NP    1024          // padded sequence length (996 -> 1024)
#define NREAL 996
#define BQ    128           // batch

// padded LDS leading dims (halves): +8 pad => lane-to-lane bank stride 4 mod 64
#define LDK   136           // 128-col rows
#define LDV   264           // 256-col rows
#define LDS_S 40            // 32-col rows (repack scratch)

__device__ __forceinline__ v8f wmma16(v16h a, v16h b, v8f c) {
  return __builtin_amdgcn_wmma_f32_16x16x32_f16(
      /*neg_a=*/false, a, /*neg_b=*/false, b,
      /*c_mod=*/(short)0, c, /*reuse_a=*/false, /*reuse_b=*/false);
}

// A-fragment K index for half h (0..15), lane half-group hi (0/1)
__device__ __forceinline__ int a_kidx(int h, int hi) {
  return ((h & 8) << 1) + hi * 8 + (h & 7);
}

// A fragment: 16x32 tile of row-major f16 matrix src (leading dim ldm).
// Per lane: two contiguous 8-half segments -> 2x (ds|global)_load_b128.
__device__ __forceinline__ v16h load_a_frag(const _Float16* src, int ldm,
                                            int row0, int k0) {
  const int lane = threadIdx.x & 31;
  const int m  = row0 + (lane & 15);
  const int hi = lane >> 4;
  v16h a;
#pragma unroll
  for (int h = 0; h < 16; ++h) a[h] = src[m * ldm + k0 + a_kidx(h, hi)];
  return a;
}

// B fragment where B[k][n] = srcT[(n0+n)*ldk + k0+k]  (contiguous in k per lane)
__device__ __forceinline__ v16h load_bT_frag(const _Float16* srcT, int ldk,
                                             int k0, int n0) {
  const int lane = threadIdx.x & 31;
  const int n  = n0 + (lane & 15);
  const int hi = lane >> 4;
  v16h b;
#pragma unroll
  for (int h = 0; h < 16; ++h) b[h] = srcT[n * ldk + k0 + a_kidx(h, hi)];
  return b;
}

// ---------------------------------------------------------------------------
// Kernel 1: x = token_emb[kmer] + sinusoidal pos * pos_scale   (pad rows = 0)
// ---------------------------------------------------------------------------
__global__ void embed_kernel(const int* __restrict__ kmer,
                             const float* __restrict__ emb,
                             const float* __restrict__ pscale,
                             float* __restrict__ x) {
  const int tid = threadIdx.x;
  const int b = blockIdx.x >> 2;
  const int t = (blockIdx.x & 3) * 256 + tid;
  const size_t tok = (size_t)b * NP + t;
  if (t >= NREAL) {
#pragma unroll
    for (int i = 0; i < 16; ++i) x[tok * 16 + i] = 0.0f;
    return;
  }
  const int id = kmer[(size_t)b * NREAL + t];
  const float ps = *pscale;
  const float ft = (float)t;
#pragma unroll
  for (int i = 0; i < 8; ++i) {
    float ang = ft * __powf(10000.0f, -(float)i / 8.0f);
    x[tok * 16 + i]     = emb[id * 16 + i]     + __sinf(ang) * ps;
    x[tok * 16 + i + 8] = emb[id * 16 + i + 8] + __cosf(ang) * ps;
  }
}

// ---------------------------------------------------------------------------
// Kernel 2 (per layer): scale-norm + token shift + Wh/Wqk projections + silu
// + per-head affine + rotary. quad_q/quad_k/lin_q row-major; lin_k and v are
// written TRANSPOSED ([b][d][t]) so the WMMA kernels stage LDS conflict-free.
// Head outputs stream straight to memory (no o[128] temp -> no spills).
// ---------------------------------------------------------------------------
__global__ void __launch_bounds__(256, 2)
proj_kernel(const float* __restrict__ x,
            const float* __restrict__ ngp,
            const float* __restrict__ Wh,
            const float* __restrict__ bh,
            const float* __restrict__ Wqk,
            const float* __restrict__ bqk,
            const float* __restrict__ gamma,
            const float* __restrict__ beta,
            _Float16* __restrict__ qq,  _Float16* __restrict__ kk,
            _Float16* __restrict__ lq,  _Float16* __restrict__ lkT,
            _Float16* __restrict__ vvT, _Float16* __restrict__ gt) {
  __shared__ float sWh[16 * 64], sbh[64], sWqk[16 * 128], sbqk[128];
  __shared__ float sG[4 * 128], sB[4 * 128];
  const int tid = threadIdx.x;
  for (int i = tid; i < 16 * 64;  i += 256) sWh[i]  = Wh[i];
  for (int i = tid; i < 16 * 128; i += 256) sWqk[i] = Wqk[i];
  if (tid < 64)  sbh[tid]  = bh[tid];
  if (tid < 128) sbqk[tid] = bqk[tid];
  for (int i = tid; i < 512; i += 256) { sG[i] = gamma[i]; sB[i] = beta[i]; }
  __syncthreads();

  const int b = blockIdx.x >> 2;
  const int t = (blockIdx.x & 3) * 256 + tid;
  const size_t tok = (size_t)b * NP + t;
  const size_t bNP = (size_t)b * NP;

  if (t >= NREAL) {  // zero padded tail so attention needs no keymask
    for (int j = 0; j < 128; ++j) {
      qq[tok * 128 + j] = (_Float16)0.f; kk[tok * 128 + j] = (_Float16)0.f;
      lq[tok * 128 + j] = (_Float16)0.f;
      lkT[(bNP * 128) + (size_t)j * NP + t] = (_Float16)0.f;
    }
    for (int j = 0; j < 32; ++j) {
      vvT[(bNP * 32) + (size_t)j * NP + t] = (_Float16)0.f;
      gt[tok * 32 + j] = (_Float16)0.f;
    }
    return;
  }

  const float g = *ngp;
  float cur[16], ss = 0.f;
#pragma unroll
  for (int i = 0; i < 16; ++i) { cur[i] = x[tok * 16 + i]; ss += cur[i] * cur[i]; }
  const float sc = g / fmaxf(sqrtf(ss) * 0.25f, 1e-5f);

  float sh[16];
#pragma unroll
  for (int i = 8; i < 16; ++i) sh[i] = cur[i] * sc;   // xp: current token
  if (t > 0) {                                        // xs: previous token
    float pv[16], ps = 0.f;
#pragma unroll
    for (int i = 0; i < 16; ++i) { pv[i] = x[(tok - 1) * 16 + i]; ps += pv[i] * pv[i]; }
    const float psc = g / fmaxf(sqrtf(ps) * 0.25f, 1e-5f);
#pragma unroll
    for (int i = 0; i < 8; ++i) sh[i] = pv[i] * psc;
  } else {
#pragma unroll
    for (int i = 0; i < 8; ++i) sh[i] = 0.f;
  }

  // hid = silu(sh @ Wh + bh) -> v (transposed) | gate
  for (int j = 0; j < 64; ++j) {
    float a = sbh[j];
#pragma unroll
    for (int i = 0; i < 16; ++i) a += sh[i] * sWh[i * 64 + j];
    float s = a / (1.0f + __expf(-a));
    if (j < 32) vvT[(bNP * 32) + (size_t)j * NP + t] = (_Float16)s;
    else        gt[tok * 32 + (j - 32)] = (_Float16)s;
  }

  // qk = silu(sh @ Wqk + bqk)
  float qkv[128];
  for (int j = 0; j < 128; ++j) {
    float a = sbqk[j];
#pragma unroll
    for (int i = 0; i < 16; ++i) a += sh[i] * sWqk[i * 128 + j];
    qkv[j] = a / (1.0f + __expf(-a));
  }

  // rotary tables for first 32 dims
  float cs[16], sn[16];
#pragma unroll
  for (int i = 0; i < 16; ++i) {
    float ang = (float)t * __powf(10000.0f, -(float)i / 16.0f);
    cs[i] = __cosf(ang); sn[i] = __sinf(ang);
  }

  // heads: 0 quad_q (row), 1 lin_q (row), 2 quad_k (row), 3 lin_k (transposed)
#pragma unroll
  for (int hh = 0; hh < 4; ++hh) {
    _Float16* rowdst = (hh == 0) ? (qq + tok * 128)
                     : (hh == 1) ? (lq + tok * 128)
                     : (hh == 2) ? (kk + tok * 128) : nullptr;
    _Float16* trdst  = lkT + bNP * 128 + t;   // + j*NP, only for hh==3
#pragma unroll
    for (int i = 0; i < 16; ++i) {
      float a  = qkv[2 * i]     * sG[hh * 128 + 2 * i]     + sB[hh * 128 + 2 * i];
      float bb = qkv[2 * i + 1] * sG[hh * 128 + 2 * i + 1] + sB[hh * 128 + 2 * i + 1];
      float e0 = a * cs[i] - bb * sn[i];
      float e1 = bb * cs[i] + a * sn[i];
      if (hh != 3) {
        rowdst[2 * i]     = (_Float16)e0;
        rowdst[2 * i + 1] = (_Float16)e1;
      } else {
        trdst[(size_t)(2 * i) * NP]     = (_Float16)e0;
        trdst[(size_t)(2 * i + 1) * NP] = (_Float16)e1;
      }
    }
    for (int j = 32; j < 128; ++j) {
      float e = qkv[j] * sG[hh * 128 + j] + sB[hh * 128 + j];
      if (hh != 3) rowdst[j] = (_Float16)e;
      else         trdst[(size_t)j * NP] = (_Float16)e;
    }
  }
}

// ---------------------------------------------------------------------------
// Kernel 3 (per layer): quadratic attention per (batch, group of 256).
// S = Q*K^T/256 ; A = relu(S)^2 ; O = A*V.  All-WMMA core.
// LDS (dynamic, ~94.5KB): K 256x(128+8) + V^T 32x(256+8) + 8x16x(32+8) repack.
// 4 B-fragments live at a time (batched loads per sub-tile) => no spills;
// the wait for sub-tile 1's loads overlaps sub-tile 0's WMMA chain.
// ---------------------------------------------------------------------------
__global__ void __launch_bounds__(256, 2)
attn_kernel(const _Float16* __restrict__ qq,
            const _Float16* __restrict__ kk,
            const _Float16* __restrict__ vvT,
            float* __restrict__ qout) {
  extern __shared__ char smem[];
  _Float16* sK  = (_Float16*)smem;              // 256 x LDK
  _Float16* sVT = sK + 256 * LDK;               // 32  x LDV  (V transposed)
  _Float16* sS  = sVT + 32 * LDV;               // 8 waves x 16 x LDS_S

  const int tid = threadIdx.x;
  const int b = blockIdx.x >> 2;
  const int gch = blockIdx.x & 3;
  const size_t base = (size_t)blockIdx.x * 256;  // token base: b*1024 + g*256

  // stage K (row-major, padded rows) and V^T -- all contiguous b128 copies
  {
    const uint4* gk = (const uint4*)(kk + base * 128);
    for (int i = tid; i < 256 * 16; i += 256) {
      int n = i >> 4, c0 = (i & 15) * 8;
      *(uint4*)(sK + n * LDK + c0) = gk[i];
    }
    const _Float16* gvt = vvT + (size_t)b * 32 * NP + gch * 256;
    for (int i = tid; i < 32 * 32; i += 256) {
      int e = i >> 5, c0 = (i & 31) * 8;
      *(uint4*)(sVT + e * LDV + c0) = *(const uint4*)(gvt + (size_t)e * NP + c0);
    }
  }
  __syncthreads();

  const int wave = tid >> 5, lane = tid & 31;
  const int hi = lane >> 4, ln = lane & 15;
  _Float16* sSw = sS + wave * 16 * LDS_S;

  const _Float16* gQ = qq + base * 128;   // Q fragments straight from global

  for (int qt = 0; qt < 2; ++qt) {               // 2 q-tiles per wave
    const int qr0 = (wave * 2 + qt) * 16;
    const v16h aQ0 = load_a_frag(gQ, 128, qr0, 0);
    const v16h aQ1 = load_a_frag(gQ, 128, qr0, 32);
    const v16h aQ2 = load_a_frag(gQ, 128, qr0, 64);
    const v16h aQ3 = load_a_frag(gQ, 128, qr0, 96);
    v8f o0 = {}, o1 = {};

    for (int j0 = 0; j0 < 8; ++j0) {             // 32 keys per chunk
      const int jb = j0 * 32;

      // ---- S sub-tile 0 (keys jb..jb+15): batch 4 loads, 4 WMMAs ----
      v16h bk0 = load_bT_frag(sK, LDK, 0,  jb);
      v16h bk1 = load_bT_frag(sK, LDK, 32, jb);
      v16h bk2 = load_bT_frag(sK, LDK, 64, jb);
      v16h bk3 = load_bT_frag(sK, LDK, 96, jb);
      v8f c0 = {};
      c0 = wmma16(aQ0, bk0, c0);
      c0 = wmma16(aQ1, bk1, c0);
      c0 = wmma16(aQ2, bk2, c0);
      c0 = wmma16(aQ3, bk3, c0);

      // ---- S sub-tile 1 (keys jb+16..jb+31): loads overlap c0's WMMAs ----
      bk0 = load_bT_frag(sK, LDK, 0,  jb + 16);
      bk1 = load_bT_frag(sK, LDK, 32, jb + 16);
      bk2 = load_bT_frag(sK, LDK, 64, jb + 16);
      bk3 = load_bT_frag(sK, LDK, 96, jb + 16);
      v8f c1 = {};
      c1 = wmma16(aQ0, bk0, c1);
      c1 = wmma16(aQ1, bk1, c1);
      c1 = wmma16(aQ2, bk2, c1);
      c1 = wmma16(aQ3, bk3, c1);

      // V fragments for this chunk (contiguous from V^T)
      const v16h bV0 = load_bT_frag(sVT, LDV, jb, 0);
      const v16h bV1 = load_bT_frag(sVT, LDV, jb, 16);

      // relu(S/256)^2 = max(S,0)^2 * 2^-16 ; repack C(f32)->A(f16) via LDS
#pragma unroll
      for (int r = 0; r < 8; ++r) {
        float m0 = fmaxf(c0[r], 0.f), m1 = fmaxf(c1[r], 0.f);
        sSw[(r + hi * 8) * LDS_S + ln]      = (_Float16)(m0 * m0 * (1.f / 65536.f));
        sSw[(r + hi * 8) * LDS_S + 16 + ln] = (_Float16)(m1 * m1 * (1.f / 65536.f));
      }
      const v16h aA = load_a_frag(sSw, LDS_S, 0, 0);
      o0 = wmma16(aA, bV0, o0);
      o1 = wmma16(aA, bV1, o1);
    }

    float* out = qout + (base + qr0) * 32;
#pragma unroll
    for (int r = 0; r < 8; ++r) {
      out[(r + hi * 8) * 32 + ln]      = o0[r];
      out[(r + hi * 8) * 32 + 16 + ln] = o1[r];
    }
  }
}

// ---------------------------------------------------------------------------
// Kernel 4 (per layer): lin_kv[b] = (lin_k^T @ V) / n  -> (128 x 32) per batch.
// M=128, N=32, K=1024 in 8 chunks; all fragments contiguous from transposed
// global copies; per K-step: batched loads then 2 WMMAs (low pressure).
// ---------------------------------------------------------------------------
__global__ void __launch_bounds__(256, 2)
linkv_kernel(const _Float16* __restrict__ lkT,
             const _Float16* __restrict__ vvT,
             float* __restrict__ lkv, float invn) {
  __shared__ _Float16 sLKT[128 * LDK];   // [d][k] rows padded
  __shared__ _Float16 sVT2[32 * LDK];    // [e][k] rows padded
  const int tid = threadIdx.x, b = blockIdx.x;
  const int wave = tid >> 5, lane = tid & 31;
  const int hi = lane >> 4, ln = lane & 15;

  v8f o0 = {}, o1 = {};
  for (int kc = 0; kc < 8; ++kc) {
    const int k0g = kc * 128;
    const _Float16* glt = lkT + (size_t)b * 128 * NP + k0g;
    for (int i = tid; i < 128 * 16; i += 256) {
      int d = i >> 4, c0 = (i & 15) * 8;
      *(uint4*)(sLKT + d * LDK + c0) = *(const uint4*)(glt + (size_t)d * NP + c0);
    }
    const _Float16* gvt = vvT + (size_t)b * 32 * NP + k0g;
    for (int i = tid; i < 32 * 16; i += 256) {
      int e = i >> 4, c0 = (i & 15) * 8;
      *(uint4*)(sVT2 + e * LDK + c0) = *(const uint4*)(gvt + (size_t)e * NP + c0);
    }
    __syncthreads();

#pragma unroll
    for (int k2 = 0; k2 < 4; ++k2) {
      const v16h aL = load_a_frag(sLKT, LDK, wave * 16, k2 * 32);
      const v16h b0 = load_bT_frag(sVT2, LDK, k2 * 32, 0);
      const v16h b1 = load_bT_frag(sVT2, LDK, k2 * 32, 16);
      o0 = wmma16(aL, b0, o0);
      o1 = wmma16(aL, b1, o1);
    }
    __syncthreads();
  }

  const int d0 = wave * 16 + hi * 8;
#pragma unroll
  for (int r = 0; r < 8; ++r) {
    lkv[(size_t)b * 4096 + (d0 + r) * 32 + ln]      = o0[r] * invn;
    lkv[(size_t)b * 4096 + (d0 + r) * 32 + 16 + ln] = o1[r] * invn;
  }
}

// ---------------------------------------------------------------------------
// Kernel 5 (per layer): lin_out = lin_q @ lin_kv ; out = gate*(quad+lin);
// x += out @ Wo + bo.   lin_kv + Wo staged in LDS.
// ---------------------------------------------------------------------------
__global__ void __launch_bounds__(256, 2)
out_kernel(float* __restrict__ x,
           const _Float16* __restrict__ lq,
           const _Float16* __restrict__ gt,
           const float* __restrict__ qout,
           const float* __restrict__ lkv,
           const float* __restrict__ Wo,
           const float* __restrict__ bo) {
  __shared__ float sKV[128 * 32];  // 16KB
  __shared__ float sWo[32 * 16];
  __shared__ float sbo[16];
  const int tid = threadIdx.x;
  const int b = blockIdx.x >> 2;
  const int t = (blockIdx.x & 3) * 256 + tid;
  for (int i = tid; i < 4096; i += 256) sKV[i] = lkv[(size_t)b * 4096 + i];
  for (int i = tid; i < 512;  i += 256) sWo[i] = Wo[i];
  if (tid < 16) sbo[tid] = bo[tid];
  __syncthreads();
  if (t >= NREAL) return;

  const size_t tok = (size_t)b * NP + t;
  float o[32];
#pragma unroll
  for (int e = 0; e < 32; ++e) o[e] = qout[tok * 32 + e];
  for (int d = 0; d < 128; ++d) {
    const float q = (float)lq[tok * 128 + d];
#pragma unroll
    for (int e = 0; e < 32; ++e) o[e] += q * sKV[d * 32 + e];
  }
#pragma unroll
  for (int e = 0; e < 32; ++e) o[e] *= (float)gt[tok * 32 + e];
#pragma unroll
  for (int i = 0; i < 16; ++i) {
    float y = sbo[i];
#pragma unroll
    for (int e = 0; e < 32; ++e) y += o[e] * sWo[e * 16 + i];
    x[tok * 16 + i] += y;   // residual
  }
}

// ---------------------------------------------------------------------------
// Kernel 6: h = max_vocab( scale_norm(x, final_g) @ Wl + bl ).
// Wl (64KB) + bl staged in dynamic LDS.
// ---------------------------------------------------------------------------
__global__ void logits_kernel(const float* __restrict__ x,
                              const float* __restrict__ fgp,
                              const float* __restrict__ Wl,
                              const float* __restrict__ bl,
                              float* __restrict__ hout) {
  extern __shared__ char smem[];
  float* sWl = (float*)smem;            // 16 x 1024
  float* sbl = sWl + 16 * 1024;         // 1024
  const int tid = threadIdx.x;
  for (int i = tid; i < 16 * 1024; i += 256) sWl[i] = Wl[i];
  for (int i = tid; i < 1024;      i += 256) sbl[i] = bl[i];
  __syncthreads();

  const int b = blockIdx.x >> 2;
  const int t = (blockIdx.x & 3) * 256 + tid;
  if (t >= NREAL) return;
  const size_t tok = (size_t)b * NP + t;

  float v[16], ss = 0.f;
#pragma unroll
  for (int i = 0; i < 16; ++i) { v[i] = x[tok * 16 + i]; ss += v[i] * v[i]; }
  const float sc = (*fgp) / fmaxf(sqrtf(ss) * 0.25f, 1e-5f);
#pragma unroll
  for (int i = 0; i < 16; ++i) v[i] *= sc;

  float m = -1e30f;
  for (int vo = 0; vo < 1024; ++vo) {
    float a = sbl[vo];
#pragma unroll
    for (int i = 0; i < 16; ++i) a += v[i] * sWl[i * 1024 + vo];
    m = fmaxf(m, a);
  }
  hout[(size_t)b * NREAL + t] = m;
}

// ---------------------------------------------------------------------------
// Kernel 7: out[b] = relu(h[b] @ W1 + b1) @ W2 + b2  (one block per batch)
// ---------------------------------------------------------------------------
__global__ void head_kernel(const float* __restrict__ h,
                            const float* __restrict__ W1,
                            const float* __restrict__ b1,
                            const float* __restrict__ W2,
                            const float* __restrict__ b2,
                            float* __restrict__ out) {
  __shared__ float part[256 * 32];  // 32KB
  __shared__ float red[32];
  const int b = blockIdx.x, tid = threadIdx.x;
  float acc[32];
#pragma unroll
  for (int j = 0; j < 32; ++j) acc[j] = 0.f;
  for (int i = tid; i < NREAL; i += 256) {
    const float hv = h[(size_t)b * NREAL + i];
#pragma unroll
    for (int j = 0; j < 32; ++j) acc[j] += hv * W1[i * 32 + j];
  }
#pragma unroll
  for (int j = 0; j < 32; ++j) part[tid * 32 + j] = acc[j];
  __syncthreads();
  if (tid < 32) {
    float s = b1[tid];
    for (int r = 0; r < 256; ++r) s += part[r * 32 + tid];
    s = fmaxf(s, 0.f);
    red[tid] = s * W2[tid];
  }
  __syncthreads();
  if (tid == 0) {
    float s = *b2;
#pragma unroll
    for (int j = 0; j < 32; ++j) s += red[j];
    out[b] = s;
  }
}

// ---------------------------------------------------------------------------
extern "C" void kernel_launch(void* const* d_in, const int* in_sizes, int n_in,
                              void* d_out, int out_size, void* d_ws, size_t ws_size,
                              hipStream_t stream) {
  const int*   kmer  = (const int*)  d_in[0];
  const float* emb   = (const float*)d_in[1];
  const float* pscal = (const float*)d_in[2];
  const float* normg = (const float*)d_in[3];
  const float* Wh    = (const float*)d_in[4];
  const float* bh    = (const float*)d_in[5];
  const float* Wqk   = (const float*)d_in[6];
  const float* bqk   = (const float*)d_in[7];
  const float* gamma = (const float*)d_in[8];
  const float* beta  = (const float*)d_in[9];
  const float* Wo    = (const float*)d_in[10];
  const float* bo    = (const float*)d_in[11];
  const float* fg    = (const float*)d_in[12];
  const float* Wl    = (const float*)d_in[13];
  const float* bl    = (const float*)d_in[14];
  const float* W1    = (const float*)d_in[15];
  const float* b1    = (const float*)d_in[16];
  const float* W2    = (const float*)d_in[17];
  const float* b2    = (const float*)d_in[18];
  float* out = (float*)d_out;

  // workspace layout
  char* ws = (char*)d_ws;
  size_t off = 0;
  auto carve = [&](size_t bytes) -> char* {
    char* p = ws + off;
    off += (bytes + 255) & ~(size_t)255;
    return p;
  };
  const size_t T = (size_t)BQ * NP;                     // 131072 padded tokens
  float*    xbuf = (float*)   carve(T * 16 * sizeof(float));
  _Float16* qq   = (_Float16*)carve(T * 128 * sizeof(_Float16));
  _Float16* kk   = (_Float16*)carve(T * 128 * sizeof(_Float16));
  _Float16* lq   = (_Float16*)carve(T * 128 * sizeof(_Float16));
  _Float16* lkT  = (_Float16*)carve(T * 128 * sizeof(_Float16));  // [b][d][t]
  _Float16* vvT  = (_Float16*)carve(T * 32 * sizeof(_Float16));   // [b][e][t]
  _Float16* gtb  = (_Float16*)carve(T * 32 * sizeof(_Float16));
  float*    qob  = (float*)   carve(T * 32 * sizeof(float));
  float*    lkv  = (float*)   carve((size_t)BQ * 128 * 32 * sizeof(float));
  float*    hbuf = (float*)   carve((size_t)BQ * NREAL * sizeof(float));
  (void)in_sizes; (void)n_in; (void)out_size; (void)ws_size;

  const dim3 blk(256);
  const dim3 gTok(BQ * (NP / 256));   // 512 blocks, one 256-token slab each

  embed_kernel<<<gTok, blk, 0, stream>>>(kmer, emb, pscal, xbuf);

  const size_t attn_lds = (size_t)(256 * LDK + 32 * LDV + 8 * 16 * LDS_S)
                          * sizeof(_Float16);           // ~94.5KB
  for (int L = 0; L < 8; ++L) {
    proj_kernel<<<gTok, blk, 0, stream>>>(
        xbuf, normg + L, Wh + (size_t)L * 16 * 64, bh + (size_t)L * 64,
        Wqk + (size_t)L * 16 * 128, bqk + (size_t)L * 128,
        gamma + (size_t)L * 4 * 128, beta + (size_t)L * 4 * 128,
        qq, kk, lq, lkT, vvT, gtb);
    attn_kernel<<<dim3(BQ * 4), blk, attn_lds, stream>>>(qq, kk, vvT, qob);
    linkv_kernel<<<dim3(BQ), blk, 0, stream>>>(lkT, vvT, lkv, 1.0f / (float)NREAL);
    out_kernel<<<gTok, blk, 0, stream>>>(
        xbuf, lq, gtb, qob, lkv, Wo + (size_t)L * 32 * 16, bo + (size_t)L * 16);
  }

  const size_t log_lds = (size_t)(16 * 1024 + 1024) * sizeof(float);  // 68KB
  logits_kernel<<<gTok, blk, log_lds, stream>>>(xbuf, fg, Wl, bl, hbuf);
  head_kernel<<<dim3(BQ), blk, 0, stream>>>(hbuf, W1, b1, W2, b2, out);
}
